// Encoder_6949257085628
// MI455X (gfx1250) — matrix operand order repaired and checked by reading, not compiled, not run on hardware
//
#include <hip/hip_runtime.h>

typedef __attribute__((ext_vector_type(16))) _Float16 v16h;
typedef __attribute__((ext_vector_type(8)))  _Float16 v8h;
typedef __attribute__((ext_vector_type(8)))  float    v8f;

#define TSTEPS  20
#define HIDN    64
#define EMBN    64
#define NGATE   256   // 4*HIDN
#define RPW     16    // batch rows per wave (one WMMA M-tile)
#define NWAVES  8
#define RPB     128   // rows per block = RPW*NWAVES
#define WSTRIDE 80    // f16 elems per W_hh row in LDS (160B, 32B-aligned, bank-spread)
#define HSTRIDE 80    // f16 elems per h row in LDS

// gfx1250 has V_TANH_F32; fall back to exp+rcp (saturates correctly: exp->inf, rcp->0)
__device__ __forceinline__ float fast_tanh(float x) {
#if __has_builtin(__builtin_amdgcn_tanhf)
  return __builtin_amdgcn_tanhf(x);
#else
  const float e = __expf(-2.f * x);                    // v_exp_f32
  return fmaf(2.f, __builtin_amdgcn_rcpf(1.f + e), -1.f);  // 2/(1+e) - 1
#endif
}
__device__ __forceinline__ float fast_sigmoid(float x) {
  return fmaf(fast_tanh(0.5f * x), 0.5f, 0.5f);
}

__global__ __launch_bounds__(256) void lstm_enc_kernel(
    const float* __restrict__ obs,     // (T, B, 2)
    const float* __restrict__ W_emb,   // (64, 2)
    const float* __restrict__ b_emb,   // (64)
    const float* __restrict__ W_ih,    // (256, 64)
    const float* __restrict__ W_hh,    // (256, 64)
    const float* __restrict__ b_ih,    // (256)
    const float* __restrict__ b_hh,    // (256)
    float* __restrict__ out,           // (B, 64) -> h_T
    int batch)
{
  __shared__ __align__(32) _Float16 sWhh[NGATE * WSTRIDE];          // 40 KB
  __shared__ __align__(32) _Float16 sH[NWAVES * RPW * HSTRIDE];     // 20 KB
  __shared__ float sObs[TSTEPS * RPB * 2];                          // 20 KB
  __shared__ float sWc[NGATE * 2];                                  // 2 KB
  __shared__ float sBias[NGATE];                                    // 1 KB

  const int  tid  = threadIdx.x;
  const long row0 = (long)blockIdx.x * RPB;

  // ---- setup: fold the rank-2 input projection into per-gate (Wc0,Wc1,bias) ----
  // x_gates[b][g] = obs_x*Wc0[g] + obs_y*Wc1[g] + bias_tot[g]
  {
    const int g = tid;                 // one gate row per thread (256 == NGATE)
    float wc0 = 0.f, wc1 = 0.f, bt = 0.f;
    #pragma unroll 8
    for (int e = 0; e < EMBN; ++e) {
      const float w = W_ih[g * EMBN + e];
      wc0 = fmaf(w, W_emb[e * 2 + 0], wc0);
      wc1 = fmaf(w, W_emb[e * 2 + 1], wc1);
      bt  = fmaf(w, b_emb[e], bt);
    }
    sWc[g * 2 + 0] = wc0;
    sWc[g * 2 + 1] = wc1;
    sBias[g] = bt + b_ih[g] + b_hh[g];
    #pragma unroll 8
    for (int k = 0; k < HIDN; ++k)     // W_hh -> f16 LDS, row-major, padded stride
      sWhh[g * WSTRIDE + k] = (_Float16)W_hh[g * HIDN + k];
  }
  // ---- stage observations for this block's rows (coalesced) ----
  for (int t = 0; t < TSTEPS; ++t)
    sObs[t * RPB * 2 + tid] = obs[((long)t * batch + row0) * 2 + tid];
  __syncthreads();

  const int wave = tid >> 5;
  const int lane = tid & 31;
  const int half = lane >> 4;          // K-half (A/B frags) / M-half (C/D frags)
  const int lm   = lane & 15;          // row (A frag) / col (B & C/D frags)
  const int wr0  = wave * RPW;
  _Float16* myH  = &sH[wave * RPW * HSTRIDE];

  // ---- explicitly hoist loop-invariant operands into registers ----
  // All of W_hh lives in VGPRs for the whole recurrence (32 x v16h = 256 VGPRs;
  // gfx1250 waves address up to 1024 VGPRs via s_set_vgpr_msb).
  v16h bfr[2][16];                     // [k-half][gate*4 + n-chunk]
  float wcx[16], wcy[16], wcb[16];     // folded input-projection coefficients
  #pragma unroll
  for (int gate = 0; gate < 4; ++gate) {
    #pragma unroll
    for (int j = 0; j < 4; ++j) {
      const int gg = gate * 64 + j * 16 + lm;   // B col n = this lane's gate row
      #pragma unroll
      for (int kh = 0; kh < 2; ++kh) {
        const v8h blo = *(const v8h*)&sWhh[gg * WSTRIDE + kh * 32 + 8 * half];
        const v8h bhi = *(const v8h*)&sWhh[gg * WSTRIDE + kh * 32 + 16 + 8 * half];
        #pragma unroll
        for (int i = 0; i < 8; ++i) { bfr[kh][gate * 4 + j][i] = blo[i]; bfr[kh][gate * 4 + j][i + 8] = bhi[i]; }
      }
      wcx[gate * 4 + j] = sWc[gg * 2 + 0];
      wcy[gate * 4 + j] = sWc[gg * 2 + 1];
      wcb[gate * 4 + j] = sBias[gg];
    }
  }

  // h_0 = 0 in A-fragment form; c_0 = 0 in D-layout registers
  v16h afrag[2];
  #pragma unroll
  for (int i = 0; i < 16; ++i) { afrag[0][i] = (_Float16)0.f; afrag[1][i] = (_Float16)0.f; }
  v8f cst[4];
  #pragma unroll
  for (int j = 0; j < 4; ++j) {
    #pragma unroll
    for (int v = 0; v < 8; ++v) cst[j][v] = 0.f;
  }

  #pragma unroll 1
  for (int t = 0; t < TSTEPS; ++t) {
    // obs for this wave's rows in D-layout (row m = v + 8*half)
    float ox[8], oy[8];
    #pragma unroll
    for (int v = 0; v < 8; ++v) {
      const int r = wr0 + v + 8 * half;
      ox[v] = sObs[(t * RPB + r) * 2 + 0];
      oy[v] = sObs[(t * RPB + r) * 2 + 1];
    }
    #pragma unroll
    for (int j = 0; j < 4; ++j) {       // n-chunk: hidden cols j*16..j*16+15
      // init accumulators with folded input contribution (i,f,g,o gate tiles)
      v8f acc[4];
      #pragma unroll
      for (int gate = 0; gate < 4; ++gate) {
        const int q = gate * 4 + j;
        #pragma unroll
        for (int v = 0; v < 8; ++v)
          acc[gate][v] = fmaf(ox[v], wcx[q], fmaf(oy[v], wcy[q], wcb[q]));
      }
      // gates += h @ W_hh^T : K=64 as two 16x16x32 WMMAs per gate tile
      #pragma unroll
      for (int kh = 0; kh < 2; ++kh) {
        #pragma unroll
        for (int gate = 0; gate < 4; ++gate) {
          acc[gate] = __builtin_amdgcn_wmma_f32_16x16x32_f16(
              false, afrag[kh], false, bfr[kh][gate * 4 + j],
              (short)0, acc[gate], false, false);
        }
      }
      // elementwise LSTM cell in f32 (v_tanh_f32 trans ops co-execute with XDL)
      #pragma unroll
      for (int v = 0; v < 8; ++v) {
        const float ig = fast_sigmoid(acc[0][v]);
        const float fg = fast_sigmoid(acc[1][v]);
        const float gv = fast_tanh(acc[2][v]);
        const float og = fast_sigmoid(acc[3][v]);
        const float cn = fmaf(fg, cst[j][v], ig * gv);
        cst[j][v] = cn;
        const float hn = og * fast_tanh(cn);
        // D-layout element (m = v + 8*half, n = j*16 + lm) -> f16 h tile in LDS
        myH[(v + 8 * half) * HSTRIDE + j * 16 + lm] = (_Float16)hn;
        if (t == TSTEPS - 1) {
          const long r = row0 + wr0 + v + 8 * half;
          out[r * HIDN + j * 16 + lm] = hn;       // final h_T, f32
        }
      }
    }
    // rebuild A fragments for next step from the wave-private h tile.
    // 16-bit A 16x32 layout: lane m holds K in [8*half,8*half+8) and [16+8*half,...)
    if (t + 1 < TSTEPS) {
      #pragma unroll
      for (int kh = 0; kh < 2; ++kh) {
        const v8h lo = *(const v8h*)&myH[lm * HSTRIDE + kh * 32 + 8 * half];
        const v8h hi = *(const v8h*)&myH[lm * HSTRIDE + kh * 32 + 16 + 8 * half];
        v16h a;
        #pragma unroll
        for (int i = 0; i < 8; ++i) { a[i] = lo[i]; a[i + 8] = hi[i]; }
        afrag[kh] = a;
      }
    }
  }
}

extern "C" void kernel_launch(void* const* d_in, const int* in_sizes, int n_in,
                              void* d_out, int out_size, void* d_ws, size_t ws_size,
                              hipStream_t stream) {
  (void)n_in; (void)out_size; (void)d_ws; (void)ws_size;
  const float* obs   = (const float*)d_in[0];
  const float* W_emb = (const float*)d_in[1];
  const float* b_emb = (const float*)d_in[2];
  const float* W_ih  = (const float*)d_in[3];
  const float* W_hh  = (const float*)d_in[4];
  const float* b_ih  = (const float*)d_in[5];
  const float* b_hh  = (const float*)d_in[6];
  float* out = (float*)d_out;

  const int batch  = in_sizes[0] / (TSTEPS * 2);   // 65536
  const int blocks = batch / RPB;                  // 512
  lstm_enc_kernel<<<blocks, 256, 0, stream>>>(obs, W_emb, b_emb, W_ih, W_hh,
                                              b_ih, b_hh, out, batch);
}
